// DeepSet_90348932039255
// MI455X (gfx1250) — compile-verified, matrix operand order checked by reading, not run on hardware
//
#include <hip/hip_runtime.h>

// ---------------------------------------------------------------------------
// CDNA5 / gfx1250 fused edge-message kernel (wave32, WMMA f16 -> f32).
// Weights pre-swizzled into WMMA B-fragment order (1KB blocks, one v16h load
// per lane). LDS stages padded (+8 halves) to kill bank conflicts on the
// row-strided A-fragment ds_load_b128s.
// ---------------------------------------------------------------------------

typedef __attribute__((ext_vector_type(16))) _Float16 v16h;
typedef __attribute__((ext_vector_type(8)))  float    v8f;

#define MT 64                 // edges per workgroup (4 M-tiles of 16)
#define STRA 72               // sAttr  row stride (64 + 8 pad)
#define STRD 136              // sDij   row stride (128 + 8 pad)
#define STRG 264              // sGamma row stride (256 + 8 pad)
#define PI_F 3.14159265358979f

__device__ inline v8f wmma_f16(v16h a, v16h b, v8f c) {
  return __builtin_amdgcn_wmma_f32_16x16x32_f16(
      /*neg_a=*/false, a, /*neg_b=*/false, b,
      /*c_mod=*/(short)0, c, /*reuse_a=*/false, /*reuse_b=*/false);
}

__device__ inline v8f zero8() {
  v8f z = {0.f, 0.f, 0.f, 0.f, 0.f, 0.f, 0.f, 0.f};
  return z;
}

// A fragment (16x32 f16) from a row-major LDS matrix (padded stride).
// Lanes 0-15 hold row M=lane with K {0..7, 16..23};
// lanes 16-31 hold row M=lane-16 with K {8..15, 24..31}.
__device__ inline v16h load_a_lds(const _Float16* p, int stride, int row0, int k0) {
  const int lane = threadIdx.x & 31;
  const int half = lane >> 4;
  const int r    = lane & 15;
  const _Float16* base = p + (row0 + r) * stride + k0 + 8 * half;
  v16h a;
#pragma unroll
  for (int i = 0; i < 8; ++i) a[i] = base[i];
#pragma unroll
  for (int i = 0; i < 8; ++i) a[8 + i] = base[16 + i];
  return a;
}

// B fragment from packed (fragment-order) weights: one 32-byte load per lane.
__device__ inline v16h load_b_pk(const _Float16* wpk, int frag) {
  const int lane = threadIdx.x & 31;
  return *(const v16h*)(wpk + ((size_t)frag * 32 + lane) * 16);
}

// Decompose packed index -> (k, n) source coordinates of the B element.
// idx layout: ((frag)*32 + lane)*16 + i, frag = nt*KT + kt.
__device__ inline void pk_coords(int idx, int KT, int& k, int& n) {
  int i    = idx & 15;
  int lane = (idx >> 4) & 31;
  int frag = idx >> 9;
  int kt   = frag % KT, nt = frag / KT;
  int h    = lane >> 4, c = lane & 15;
  int kk   = (i < 8 ? i : i + 8) + 8 * h;
  k = kt * 32 + kk;
  n = nt * 16 + c;
}

// ---------------------------------------------------------------------------
// Prep kernels: weight folding + f16 conversion + fragment swizzle.
// ---------------------------------------------------------------------------

// Wfold[n][k] = sum_h W_dij[n][h] * W_dp[h][k]  (n<128, k<64, h<256), packed.
// bfold[n]   = sum_h W_dij[n][h] * b_dp[h]
__global__ void prep_fold(const float* __restrict__ W_dij, const float* __restrict__ W_dp,
                          const float* __restrict__ b_dp,
                          _Float16* __restrict__ WfoldP, float* __restrict__ bfold) {
  int idx = blockIdx.x * 256 + threadIdx.x;
  if (idx < 128 * 64) {              // KT=2, NT=8 -> 16 frags * 512
    int k, n;
    pk_coords(idx, 2, k, n);
    float acc = 0.f;
    for (int h = 0; h < 256; ++h) acc += W_dij[n * 259 + h] * W_dp[h * 64 + k];
    WfoldP[idx] = (_Float16)acc;
  } else if (idx < 128 * 64 + 128) {
    int o = idx - 128 * 64;
    float acc = 0.f;
    for (int h = 0; h < 256; ++h) acc += W_dij[o * 259 + h] * b_dp[h];
    bfold[o] = acc;
  }
}

// A_I = emb @ W_ai^T + b_ai (100x128); A_J = emb @ W_aj^T + b_aj (100x256)
__global__ void prep_atom(const float* __restrict__ emb,
                          const float* __restrict__ W_ai, const float* __restrict__ b_ai,
                          const float* __restrict__ W_aj, const float* __restrict__ b_aj,
                          float* __restrict__ A_I, float* __restrict__ A_J) {
  int idx = blockIdx.x * 256 + threadIdx.x;
  if (idx < 100 * 128) {
    int t = idx >> 7, c = idx & 127;
    float acc = b_ai[c];
    for (int k = 0; k < 128; ++k) acc += emb[t * 128 + k] * W_ai[c * 128 + k];
    A_I[idx] = acc;
  } else if (idx < 100 * 128 + 100 * 256) {
    int j = idx - 100 * 128;
    int t = j >> 8, c = j & 255;
    float acc = b_aj[c];
    for (int k = 0; k < 128; ++k) acc += emb[t * 128 + k] * W_aj[c * 128 + k];
    A_J[j] = acc;
  }
}

// G1 = A_I @ Wg[:, 0:128]^T + b_gamma ; G2 = A_J @ Wg[:, 128:384]^T  (each 100x256)
__global__ void prep_g(const float* __restrict__ A_I, const float* __restrict__ A_J,
                       const float* __restrict__ W_gamma, const float* __restrict__ b_gamma,
                       float* __restrict__ G1, float* __restrict__ G2) {
  int idx = blockIdx.x * 256 + threadIdx.x;
  if (idx < 100 * 256) {
    int t = idx >> 8, n = idx & 255;
    float acc = b_gamma[n];
    for (int c = 0; c < 128; ++c) acc += A_I[t * 128 + c] * W_gamma[n * 512 + c];
    G1[idx] = acc;
  } else if (idx < 2 * 100 * 256) {
    int j = idx - 100 * 256;
    int t = j >> 8, n = j & 255;
    float acc = 0.f;
    for (int c = 0; c < 256; ++c) acc += A_J[t * 256 + c] * W_gamma[n * 512 + 128 + c];
    G2[j] = acc;
  }
}

// Wg3 packed: K=128 (KT=4), N=256 (NT=16); src = W_gamma[n][384+k]
// Wexp packed: per expert g, K=256 (KT=8), N=128 (NT=8); src = W_exp[g][n][k]
__global__ void prep_wt(const float* __restrict__ W_gamma, const float* __restrict__ W_exp,
                        _Float16* __restrict__ Wg3P, _Float16* __restrict__ WexpP) {
  int idx = blockIdx.x * 256 + threadIdx.x;
  if (idx < 128 * 256) {
    int k, n;
    pk_coords(idx, 4, k, n);
    Wg3P[idx] = (_Float16)W_gamma[n * 512 + 384 + k];
  } else if (idx < 128 * 256 + 10 * 256 * 128) {
    int j = idx - 128 * 256;
    int g = j >> 15;            // 32768 halves per expert
    int r = j & 32767;
    int k, n;
    pk_coords(r, 8, k, n);
    WexpP[j] = (_Float16)W_exp[((size_t)g * 128 + n) * 256 + k];
  }
}

// Zero atom_x/vec accumulators and emit z/pos/batch pass-through tail.
__global__ void init_out(float* __restrict__ out, const int* __restrict__ z,
                         const float* __restrict__ pos, const int* __restrict__ batch) {
  const long long A  = 40000LL * 128;   //  5,120,000  atom_x
  const long long V  = 40000LL * 384;   // 15,360,000  vec
  const long long Z0 = A + V;           // 20,480,000
  const long long P0 = Z0 + 40000;      // 20,520,000
  const long long B0 = P0 + 120000;     // 20,640,000
  const long long TOT = B0 + 40000;     // 20,680,000
  for (long long i = (long long)blockIdx.x * 256 + threadIdx.x; i < TOT;
       i += (long long)gridDim.x * 256) {
    if (i < Z0)      out[i] = 0.f;
    else if (i < P0) out[i] = (float)z[i - Z0];
    else if (i < B0) out[i] = pos[i - P0];
    else             out[i] = (float)batch[i - B0];
  }
}

// ---------------------------------------------------------------------------
// Fused per-edge kernel: 64 edges / workgroup, 8 waves (wave32).
// ---------------------------------------------------------------------------
__global__ __launch_bounds__(256) void fused_edge_kernel(
    const int* __restrict__ z, const float* __restrict__ pos,
    const int* __restrict__ src, const int* __restrict__ dst,
    const _Float16* __restrict__ WfoldP, const float* __restrict__ bfold,
    const float* __restrict__ W_dij, const float* __restrict__ b_dij,
    const float* __restrict__ G1, const float* __restrict__ G2,
    const _Float16* __restrict__ Wg3P, const _Float16* __restrict__ WexpP,
    const float* __restrict__ b_exp, const float* __restrict__ t_params,
    float* __restrict__ atom_x, float* __restrict__ vec_out, int n_edges) {
  __shared__ _Float16 sAttr[MT * STRA];    // C-scaled RBF features
  __shared__ _Float16 sDij[MT * STRD];     // d_ij_t (f16)
  __shared__ _Float16 sGamma[MT * STRG];   // gamma (f16)
  __shared__ float  sGate[MT * 10];
  __shared__ float4 sScal[MT];             // {C, ew^3, sqrt(ew), ew}
  __shared__ float  sVec[MT * 3];
  __shared__ int    sSrc[MT], sZs[MT], sZd[MT];

  const int tid  = threadIdx.x;
  const int wv   = tid >> 5;
  const int lane = tid & 31;
  const int half = lane >> 4;
  const int lcol = lane & 15;
  const int e0   = blockIdx.x * MT;

  // ---- Phase 1a: per-edge scalars + gates (threads 0..63) ----
  if (tid < MT) {
    int  e     = e0 + tid;
    bool valid = (e < n_edges);
    int  s = valid ? src[e] : 0;
    int  d = valid ? dst[e] : 0;
    float dx = pos[3 * s + 0] - pos[3 * d + 0];
    float dy = pos[3 * s + 1] - pos[3 * d + 1];
    float dz = pos[3 * s + 2] - pos[3 * d + 2];
    float w  = sqrtf(dx * dx + dy * dy + dz * dz);
    float iw = 1.f / w;
    sVec[tid * 3 + 0] = dx * iw;
    sVec[tid * 3 + 1] = dy * iw;
    sVec[tid * 3 + 2] = dz * iw;
    float ew3 = w * w * w;
    float C = (w < 5.0f) ? 0.5f * (__cosf(w * (PI_F / 5.0f)) + 1.0f) : 0.f;
    sScal[tid] = make_float4(C, ew3, sqrtf(w), w);
    sSrc[tid] = s;
    sZs[tid]  = z[s];
    sZd[tid]  = z[d];
    // distance gating softmax (10 experts)
    float iv[10];
    float mx = -1e30f;
#pragma unroll
    for (int g = 0; g < 10; ++g) {
      float a = fmaxf(fabsf(w - t_params[g]), 1e-8f);
      iv[g] = 1.f / a;
      mx = fmaxf(mx, iv[g]);
    }
    float ssum = 0.f;
#pragma unroll
    for (int g = 0; g < 10; ++g) { iv[g] = __expf(iv[g] - mx); ssum += iv[g]; }
    float inv_s = valid ? (1.f / ssum) : 0.f;  // invalid edges -> all-zero output
#pragma unroll
    for (int g = 0; g < 10; ++g) sGate[tid * 10 + g] = iv[g] * inv_s;
  }
  __syncthreads();

  // ---- Phase 1b: RBF features, pre-scaled by cutoff C ----
  const float delta = 5.0f / 31.0f;
  const float coeff = -0.5f / (delta * delta);
  for (int idx = tid; idx < MT * 64; idx += 256) {
    int    e  = idx >> 6, c = idx & 63;
    float4 sc = sScal[e];
    float  dv = (c < 32) ? sc.w : sc.y;
    float  t  = dv - (float)(c & 31) * delta;
    sAttr[e * STRA + c] = (_Float16)(__expf(coeff * t * t) * sc.x);
  }
  __syncthreads();

  // ---- Phase 2: d_ij_t = (C*attr) @ Wfold^T + C*bfold + scalar cols + b_dij ----
  {
    const int n0 = wv * 16;  // 8 waves cover N=128  (KT=2)
    v8f acc[4] = {zero8(), zero8(), zero8(), zero8()};
#pragma unroll
    for (int kk = 0; kk < 2; ++kk) {
      v16h b = load_b_pk(WfoldP, wv * 2 + kk);
#pragma unroll
      for (int m = 0; m < 4; ++m) {
        v16h a = load_a_lds(sAttr, STRA, m * 16, kk * 32);
        acc[m] = wmma_f16(a, b, acc[m]);
      }
    }
    const int   col = n0 + lcol;
    const float wc0 = W_dij[col * 259 + 256];  // ew^3 column
    const float wc1 = W_dij[col * 259 + 257];  // sqrt(ew) column
    const float wc2 = W_dij[col * 259 + 258];  // ew column
    const float bd  = b_dij[col];
    const float bf  = bfold[col];
#pragma unroll
    for (int m = 0; m < 4; ++m) {
      const int     rb = m * 16 + 8 * half;
      const float4* sp = &sScal[rb];
      _Float16*     pb = &sDij[rb * STRD + col];
#pragma unroll
      for (int i = 0; i < 8; ++i) {
        float4 sc = sp[i];
        pb[i * STRD] = (_Float16)(acc[m][i] + sc.x * bf + sc.y * wc0 +
                                  sc.z * wc1 + sc.w * wc2 + bd);
      }
    }
  }
  __syncthreads();

  // ---- Phase 3: gamma = G1[z_src] + G2[z_dst] + d_ij_t @ Wg3^T ----
#pragma unroll
  for (int nt = 0; nt < 2; ++nt) {
    const int ncol0 = (wv + 8 * nt) * 16;  // 8 waves x 2 cover N=256 (KT=4)
    v8f acc[4] = {zero8(), zero8(), zero8(), zero8()};
#pragma unroll
    for (int kk = 0; kk < 4; ++kk) {
      v16h b = load_b_pk(Wg3P, (wv + 8 * nt) * 4 + kk);
#pragma unroll
      for (int m = 0; m < 4; ++m) {
        v16h a = load_a_lds(sDij, STRD, m * 16, kk * 32);
        acc[m] = wmma_f16(a, b, acc[m]);
      }
    }
    const int col = ncol0 + lcol;
#pragma unroll
    for (int m = 0; m < 4; ++m) {
      const int  rb = m * 16 + 8 * half;
      const int* zs = &sZs[rb];
      const int* zd = &sZd[rb];
      _Float16*  pb = &sGamma[rb * STRG + col];
#pragma unroll
      for (int i = 0; i < 8; ++i) {
        float v = acc[m][i] + G1[zs[i] * 256 + col] + G2[zd[i] * 256 + col];
        pb[i * STRG] = (_Float16)v;
      }
    }
  }
  __syncthreads();

  // ---- Phase 4: experts + gate mixing + scatter ----
  {
    const int n0  = wv * 16;  // 8 waves cover expert N=128 (KT=8)
    const int col = n0 + lcol;
    float be[10];
#pragma unroll
    for (int g = 0; g < 10; ++g) be[g] = b_exp[g * 128 + col];

    v8f fin[4];
#pragma unroll
    for (int m = 0; m < 4; ++m) {
      const float* gp = &sGate[(m * 16 + 8 * half) * 10];
#pragma unroll
      for (int i = 0; i < 8; ++i) {
        float v = 0.f;
#pragma unroll
        for (int g = 0; g < 10; ++g) v += gp[i * 10 + g] * be[g];
        fin[m][i] = v;
      }
    }

#pragma unroll 1
    for (int g = 0; g < 10; ++g) {
      v8f acc[4] = {zero8(), zero8(), zero8(), zero8()};
#pragma unroll
      for (int kk = 0; kk < 8; ++kk) {
        if (g < 9)  // prefetch same fragment of the next expert (1KB, lane-spread)
          __builtin_prefetch(
              (const void*)(WexpP + (((size_t)(g + 1) * 8 + wv) * 8 + kk) * 512 +
                            lane * 16), 0, 1);
        v16h b = load_b_pk(WexpP, ((g * 8 + wv) * 8 + kk));
#pragma unroll
        for (int m = 0; m < 4; ++m) {
          v16h a = load_a_lds(sGamma, STRG, m * 16, kk * 32);
          acc[m] = wmma_f16(a, b, acc[m]);
        }
      }
#pragma unroll
      for (int m = 0; m < 4; ++m) {
        const float* gp = &sGate[(m * 16 + 8 * half) * 10 + g];
#pragma unroll
        for (int i = 0; i < 8; ++i) fin[m][i] += gp[i * 10] * acc[m][i];
      }
    }

    // scatter-add into atom_x and vec (invalid edges contribute exact zeros)
#pragma unroll
    for (int m = 0; m < 4; ++m) {
      const int    rb = m * 16 + 8 * half;
      const int*   sp = &sSrc[rb];
      const float* vp = &sVec[rb * 3];
#pragma unroll
      for (int i = 0; i < 8; ++i) {
        float v  = fin[m][i];
        int   sa = sp[i];
        unsafeAtomicAdd(atom_x + (size_t)sa * 128 + col, v);
        unsafeAtomicAdd(vec_out + ((size_t)sa * 3 + 0) * 128 + col, vp[i * 3 + 0] * v);
        unsafeAtomicAdd(vec_out + ((size_t)sa * 3 + 1) * 128 + col, vp[i * 3 + 1] * v);
        unsafeAtomicAdd(vec_out + ((size_t)sa * 3 + 2) * 128 + col, vp[i * 3 + 2] * v);
      }
    }
  }
}

// ---------------------------------------------------------------------------
// Launch
// ---------------------------------------------------------------------------
extern "C" void kernel_launch(void* const* d_in, const int* in_sizes, int n_in,
                              void* d_out, int out_size, void* d_ws, size_t ws_size,
                              hipStream_t stream) {
  (void)n_in; (void)out_size; (void)ws_size;
  const int*   z       = (const int*)  d_in[0];
  const float* pos     = (const float*)d_in[1];
  const int*   batch   = (const int*)  d_in[2];
  const int*   eidx    = (const int*)  d_in[3];
  const float* emb     = (const float*)d_in[4];
  const float* W_dp    = (const float*)d_in[5];
  const float* b_dp    = (const float*)d_in[6];
  const float* W_dij   = (const float*)d_in[7];
  const float* b_dij   = (const float*)d_in[8];
  const float* W_ai    = (const float*)d_in[9];
  const float* b_ai    = (const float*)d_in[10];
  const float* W_aj    = (const float*)d_in[11];
  const float* b_aj    = (const float*)d_in[12];
  const float* W_gamma = (const float*)d_in[13];
  const float* b_gamma = (const float*)d_in[14];
  const float* t_par   = (const float*)d_in[15];
  const float* W_exp   = (const float*)d_in[16];
  const float* b_exp   = (const float*)d_in[17];

  const int  n_edges = in_sizes[3] / 2;
  const int* src     = eidx;
  const int* dst     = eidx + n_edges;

  char*  ws  = (char*)d_ws;
  size_t off = 0;
  auto alloc = [&](size_t bytes) {
    size_t o = off;
    off += (bytes + 255) & ~(size_t)255;
    return o;
  };
  _Float16* WfoldP = (_Float16*)(ws + alloc(128 * 64 * 2));
  float*    bfold  = (float*)   (ws + alloc(128 * 4));
  float*    A_I    = (float*)   (ws + alloc(100 * 128 * 4));
  float*    A_J    = (float*)   (ws + alloc(100 * 256 * 4));
  float*    G1     = (float*)   (ws + alloc(100 * 256 * 4));
  float*    G2     = (float*)   (ws + alloc(100 * 256 * 4));
  _Float16* Wg3P   = (_Float16*)(ws + alloc(128 * 256 * 2));
  _Float16* WexpP  = (_Float16*)(ws + alloc(10 * 256 * 128 * 2));

  float* atom_x = (float*)d_out;
  float* vec    = atom_x + 40000LL * 128;

  prep_fold<<<(128 * 64 + 128 + 255) / 256, 256, 0, stream>>>(W_dij, W_dp, b_dp, WfoldP, bfold);
  prep_atom<<<(100 * 128 + 100 * 256 + 255) / 256, 256, 0, stream>>>(emb, W_ai, b_ai, W_aj, b_aj, A_I, A_J);
  prep_g<<<(2 * 100 * 256 + 255) / 256, 256, 0, stream>>>(A_I, A_J, W_gamma, b_gamma, G1, G2);
  prep_wt<<<(128 * 256 + 10 * 256 * 128 + 255) / 256, 256, 0, stream>>>(W_gamma, W_exp, Wg3P, WexpP);
  init_out<<<4096, 256, 0, stream>>>((float*)d_out, z, pos, batch);

  fused_edge_kernel<<<(n_edges + MT - 1) / MT, 256, 0, stream>>>(
      z, pos, src, dst, WfoldP, bfold, W_dij, b_dij, G1, G2, Wg3P, WexpP,
      b_exp, t_par, atom_x, vec, n_edges);
}
